// MyMultiheadAttention_62938450755676
// MI455X (gfx1250) — matrix operand order repaired and checked by reading, not compiled, not run on hardware
//
#include <hip/hip_runtime.h>

#define BATCH_ 2
#define SEQ_   2048
#define EMB_   1024
#define HEADS_ 16
#define HD_    64
#define M_TOT  (BATCH_*SEQ_)     // 4096
#define BH_    (BATCH_*HEADS_)   // 32

typedef __bf16 bf16_t;
typedef __attribute__((ext_vector_type(16))) __bf16 v16bf;
typedef __attribute__((ext_vector_type(8)))  __bf16 v8bf;
typedef __attribute__((ext_vector_type(8)))  float  v8f;

#define SHUF16(lo, hi) __builtin_shufflevector(lo, hi, 0,1,2,3,4,5,6,7,8,9,10,11,12,13,14,15)

static __device__ __forceinline__ v8f wmma_bf16(v16bf a, v16bf b, v8f c) {
  // D = A(16x32 bf16) * B(32x16 bf16) + C(16x16 f32)
  return __builtin_amdgcn_wmma_f32_16x16x32_bf16(false, a, false, b, (short)0, c, false, false);
}

// ---- 16-lane butterfly reductions on the VALU via DPP16 (no DS traffic) ----
template<int CTRL>
static __device__ __forceinline__ float dpp_mov(float x) {
  return __builtin_bit_cast(float,
      __builtin_amdgcn_update_dpp(0, __builtin_bit_cast(int, x),
                                  CTRL, 0xf, 0xf, true));
}
static __device__ __forceinline__ float red16_max(float x) {
  x = fmaxf(x, dpp_mov<0xB1>(x));    // quad_perm(1,0,3,2)  : xor1
  x = fmaxf(x, dpp_mov<0x4E>(x));    // quad_perm(2,3,0,1)  : xor2
  x = fmaxf(x, dpp_mov<0x141>(x));   // row_half_mirror     : xor4 (after 1,2)
  x = fmaxf(x, dpp_mov<0x140>(x));   // row_mirror          : xor8 (after 1,2,4)
  return x;
}
static __device__ __forceinline__ float red16_sum(float x) {
  x += dpp_mov<0xB1>(x);
  x += dpp_mov<0x4E>(x);
  x += dpp_mov<0x141>(x);
  x += dpp_mov<0x140>(x);
  return x;
}

// ---------------------------------------------------------------- cast kernel
__global__ __launch_bounds__(256)
void cvt_f32_bf16(const float* __restrict__ in, bf16_t* __restrict__ out, int n) {
  int i = blockIdx.x * 256 + threadIdx.x;
  if (i < n) out[i] = (bf16_t)in[i];
}

// ------------------------------------------------------- 128x128 WMMA GEMM
// C[M,N] = A[M,K] * Bw[N,K]^T + bias   (A, Bw row-major bf16)
// EPI==0: write f32 to Cf.   EPI==1: scatter QKV (Q scaled 0.125, V transposed)
template<int EPI, int N, int K>
__global__ __launch_bounds__(256)
void gemm_bf16_128x128(const bf16_t* __restrict__ A, const bf16_t* __restrict__ Bw,
                       const float* __restrict__ bias, float* __restrict__ Cf,
                       bf16_t* __restrict__ qo, bf16_t* __restrict__ ko,
                       bf16_t* __restrict__ vo) {
  __shared__ bf16_t As[128][40];   // +8 pad: rows 80B apart, 16B aligned
  __shared__ bf16_t Bs[128][40];

  const int tid  = threadIdx.x;
  const int wid  = tid >> 5, lane = tid & 31;
  const int wm   = wid >> 2, wn = wid & 3;     // 2x4 wave grid -> 64x32 per wave
  const int lr   = lane & 15, hs = lane >> 4;
  const int mBase = blockIdx.y * 128;
  const int nBase = blockIdx.x * 128;

  const int li = tid >> 1;          // LDS fill: row 0..127
  const int lh = (tid & 1) * 16;    // col 0 or 16

  const bf16_t* aptr = A  + (size_t)(mBase + li) * K + lh;
  const bf16_t* bptr = Bw + (size_t)(nBase + li) * K + lh;

  v8f acc[4][2] = {};

  for (int kk = 0; kk < K; kk += 32) {
    __syncthreads();
    v8bf a0v = *(const v8bf*)(aptr + kk);
    v8bf a1v = *(const v8bf*)(aptr + kk + 8);
    v8bf b0v = *(const v8bf*)(bptr + kk);
    v8bf b1v = *(const v8bf*)(bptr + kk + 8);
    *(v8bf*)&As[li][lh]     = a0v;
    *(v8bf*)&As[li][lh + 8] = a1v;
    *(v8bf*)&Bs[li][lh]     = b0v;
    *(v8bf*)&Bs[li][lh + 8] = b1v;
    if (kk + 32 < K) {
      __builtin_prefetch(aptr + kk + 32, 0, 1);   // global_prefetch_b8
      __builtin_prefetch(bptr + kk + 32, 0, 1);
    }
    __syncthreads();

    v16bf af[4], bfg[2];
#pragma unroll
    for (int mt = 0; mt < 4; ++mt) {
      int row = wm * 64 + mt * 16 + lr;
      v8bf lo = *(const v8bf*)&As[row][hs * 8];        // K = kb..kb+7
      v8bf hi = *(const v8bf*)&As[row][hs * 8 + 16];   // K = kb+16..kb+23
      af[mt] = SHUF16(lo, hi);
    }
#pragma unroll
    for (int nt = 0; nt < 2; ++nt) {
      int c = wn * 32 + nt * 16 + lr;
      v8bf lo = *(const v8bf*)&Bs[c][hs * 8];           // K = kb..kb+7
      v8bf hi = *(const v8bf*)&Bs[c][hs * 8 + 16];      // K = kb+16..kb+23
      bfg[nt] = SHUF16(lo, hi);
    }
#pragma unroll
    for (int mt = 0; mt < 4; ++mt)
#pragma unroll
      for (int nt = 0; nt < 2; ++nt)
        acc[mt][nt] = wmma_bf16(af[mt], bfg[nt], acc[mt][nt]);
  }

#pragma unroll
  for (int mt = 0; mt < 4; ++mt) {
#pragma unroll
    for (int nt = 0; nt < 2; ++nt) {
      int n = nBase + wn * 32 + nt * 16 + lr;
      float bv = bias[n];
      if (EPI == 0) {
#pragma unroll
        for (int r = 0; r < 8; ++r) {
          int row = mBase + wm * 64 + mt * 16 + r + hs * 8;
          Cf[(size_t)row * N + n] = acc[mt][nt][r] + bv;
        }
      } else {
        int which = n >> 10;
        int h = (n & 1023) >> 6;
        int d = n & 63;
#pragma unroll
        for (int r = 0; r < 8; ++r) {
          int row = mBase + wm * 64 + mt * 16 + r + hs * 8;
          int bb = row >> 11;
          int s  = row & (SEQ_ - 1);
          int bh = bb * HEADS_ + h;
          float v = acc[mt][nt][r] + bv;
          if (which == 0)      qo[((size_t)bh * SEQ_ + s) * HD_ + d] = (bf16_t)(v * 0.125f);
          else if (which == 1) ko[((size_t)bh * SEQ_ + s) * HD_ + d] = (bf16_t)v;
          else                 vo[((size_t)bh * HD_ + d) * SEQ_ + s] = (bf16_t)v; // V^T
        }
      }
    }
  }
}

// ---------------------------------------------------------- flash attention
// Block = 4 waves, same head, 4 consecutive 16-query tiles. K/V^T chunks of
// 64 keys staged in LDS once per block; online softmax every 64 keys.
__global__ __launch_bounds__(128)
void flash_attn(const bf16_t* __restrict__ Q, const bf16_t* __restrict__ Km,
                const bf16_t* __restrict__ Vt, bf16_t* __restrict__ Ao) {
  __shared__ bf16_t Ks[64][72];      // K chunk  : [key][d]
  __shared__ bf16_t Vs[64][72];      // V^T chunk: [d][key]
  __shared__ bf16_t Pl[4][16][72];   // per-wave P tile bounce

  const int tid = threadIdx.x;
  const int wid = tid >> 5, lane = tid & 31;
  const int lr = lane & 15, hs = lane >> 4;
  const int bh = blockIdx.y;
  const int qbase = (blockIdx.x * 4 + wid) * 16;

  // Q fragments (d 0..31, 32..63); Q pre-scaled by D^-0.5
  const bf16_t* qp = Q + ((size_t)bh * SEQ_ + qbase + lr) * HD_;
  v16bf a0, a1;
  {
    int kb = hs * 8;
    v8bf lo = *(const v8bf*)(qp + kb);
    v8bf hi = *(const v8bf*)(qp + kb + 16);
    a0 = SHUF16(lo, hi);
    lo = *(const v8bf*)(qp + 32 + kb);
    hi = *(const v8bf*)(qp + 32 + kb + 16);
    a1 = SHUF16(lo, hi);
  }

  v8f o0 = {}, o1 = {}, o2 = {}, o3 = {};
  float mrun[8], lrun[8];
#pragma unroll
  for (int r = 0; r < 8; ++r) { mrun[r] = -__builtin_inff(); lrun[r] = 0.f; }

  const bf16_t* kbp = Km + (size_t)bh * SEQ_ * HD_;
  const bf16_t* vbp = Vt + (size_t)bh * HD_ * SEQ_;

  const int fr = tid >> 1;          // LDS fill: row 0..63
  const int fc = (tid & 1) * 32;    // col 0 or 32

  for (int kc = 0; kc < SEQ_; kc += 64) {
    __syncthreads();
    {
      const bf16_t* ksrc = kbp + (size_t)(kc + fr) * HD_ + fc;
      const bf16_t* vsrc = vbp + (size_t)fr * SEQ_ + kc + fc;
      v8bf k0 = *(const v8bf*)(ksrc);
      v8bf k1 = *(const v8bf*)(ksrc + 8);
      v8bf k2 = *(const v8bf*)(ksrc + 16);
      v8bf k3 = *(const v8bf*)(ksrc + 24);
      v8bf w0 = *(const v8bf*)(vsrc);
      v8bf w1 = *(const v8bf*)(vsrc + 8);
      v8bf w2 = *(const v8bf*)(vsrc + 16);
      v8bf w3 = *(const v8bf*)(vsrc + 24);
      *(v8bf*)&Ks[fr][fc]      = k0;
      *(v8bf*)&Ks[fr][fc + 8]  = k1;
      *(v8bf*)&Ks[fr][fc + 16] = k2;
      *(v8bf*)&Ks[fr][fc + 24] = k3;
      *(v8bf*)&Vs[fr][fc]      = w0;
      *(v8bf*)&Vs[fr][fc + 8]  = w1;
      *(v8bf*)&Vs[fr][fc + 16] = w2;
      *(v8bf*)&Vs[fr][fc + 24] = w3;
      if (kc + 64 < SEQ_) {
        __builtin_prefetch(ksrc + 64 * HD_, 0, 1);
        __builtin_prefetch(vsrc + 64, 0, 1);
      }
    }
    __syncthreads();

    // S = Q K^T : 4 key-tiles x 2 d-steps
    v8f st[4];
#pragma unroll
    for (int t = 0; t < 4; ++t) {
      int krow = t * 16 + lr;
      v8bf lo0 = *(const v8bf*)&Ks[krow][hs * 8];
      v8bf hi0 = *(const v8bf*)&Ks[krow][hs * 8 + 16];
      v8bf lo1 = *(const v8bf*)&Ks[krow][32 + hs * 8];
      v8bf hi1 = *(const v8bf*)&Ks[krow][32 + hs * 8 + 16];
      v8f s = {};
      s = wmma_bf16(a0, SHUF16(lo0, hi0), s);
      s = wmma_bf16(a1, SHUF16(lo1, hi1), s);
      st[t] = s;
    }

    // online softmax: each C-tile row lives across 16 lanes of a half-wave
    float alpha[8];
#pragma unroll
    for (int r = 0; r < 8; ++r) {
      float rm = fmaxf(fmaxf(st[0][r], st[1][r]), fmaxf(st[2][r], st[3][r]));
      rm = red16_max(rm);
      float mn = fmaxf(mrun[r], rm);
      alpha[r] = __expf(mrun[r] - mn);
      mrun[r] = mn;
      float rs = 0.f;
#pragma unroll
      for (int t = 0; t < 4; ++t) {
        float p = __expf(st[t][r] - mn);
        st[t][r] = p;
        rs += p;
      }
      rs = red16_sum(rs);
      lrun[r] = lrun[r] * alpha[r] + rs;
      o0[r] *= alpha[r]; o1[r] *= alpha[r]; o2[r] *= alpha[r]; o3[r] *= alpha[r];
#pragma unroll
      for (int t = 0; t < 4; ++t)
        Pl[wid][r + hs * 8][t * 16 + lr] = (bf16_t)st[t][r];
    }
    asm volatile("s_wait_dscnt 0" ::: "memory");   // wave-local LDS RAW

    // P as two A-fragments (keys 0..31, 32..63)
    v16bf pa0, pa1;
    {
      int kb = hs * 8;
      v8bf lo = *(const v8bf*)&Pl[wid][lr][kb];
      v8bf hi = *(const v8bf*)&Pl[wid][lr][kb + 16];
      pa0 = SHUF16(lo, hi);
      lo = *(const v8bf*)&Pl[wid][lr][32 + kb];
      hi = *(const v8bf*)&Pl[wid][lr][32 + kb + 16];
      pa1 = SHUF16(lo, hi);
    }
    // O += P V : 4 d-tiles x 2 key-steps; B-frags from Vs[d][key]
#pragma unroll
    for (int nt = 0; nt < 4; ++nt) {
      int drow = nt * 16 + lr;
      v8bf lo0 = *(const v8bf*)&Vs[drow][hs * 8];
      v8bf hi0 = *(const v8bf*)&Vs[drow][hs * 8 + 16];
      v8bf lo1 = *(const v8bf*)&Vs[drow][32 + hs * 8];
      v8bf hi1 = *(const v8bf*)&Vs[drow][32 + hs * 8 + 16];
      v8f* op = (nt == 0) ? &o0 : (nt == 1) ? &o1 : (nt == 2) ? &o2 : &o3;
      *op = wmma_bf16(pa0, SHUF16(lo0, hi0), *op);
      *op = wmma_bf16(pa1, SHUF16(lo1, hi1), *op);
    }
  }

  const int bb = bh >> 4, h = bh & 15;
  bf16_t* orow = Ao + (size_t)(bb * SEQ_ + qbase) * EMB_ + h * HD_;
#pragma unroll
  for (int r = 0; r < 8; ++r) {
    float inv = 1.0f / lrun[r];
    size_t ro = (size_t)(r + hs * 8) * EMB_;
    orow[ro + 0  + lr] = (bf16_t)(o0[r] * inv);
    orow[ro + 16 + lr] = (bf16_t)(o1[r] * inv);
    orow[ro + 32 + lr] = (bf16_t)(o2[r] * inv);
    orow[ro + 48 + lr] = (bf16_t)(o3[r] * inv);
  }
}

// ----------------------------------------------------------------- launcher
extern "C" void kernel_launch(void* const* d_in, const int* in_sizes, int n_in,
                              void* d_out, int out_size, void* d_ws, size_t ws_size,
                              hipStream_t stream) {
  const float* x    = (const float*)d_in[0];
  const float* wqkv = (const float*)d_in[1];
  const float* bqkv = (const float*)d_in[2];
  const float* wout = (const float*)d_in[3];
  const float* bout = (const float*)d_in[4];
  float* out = (float*)d_out;

  char* ws = (char*)d_ws;
  size_t off = 0;
  bf16_t* xb  = (bf16_t*)(ws + off); off += (size_t)M_TOT * EMB_ * 2;      // 8 MB
  bf16_t* wqb = (bf16_t*)(ws + off); off += (size_t)3 * EMB_ * EMB_ * 2;   // 6 MB
  bf16_t* wob = (bf16_t*)(ws + off); off += (size_t)EMB_ * EMB_ * 2;       // 2 MB
  bf16_t* qb  = (bf16_t*)(ws + off); off += (size_t)BH_ * SEQ_ * HD_ * 2;  // 8 MB
  bf16_t* kb  = (bf16_t*)(ws + off); off += (size_t)BH_ * SEQ_ * HD_ * 2;  // 8 MB
  bf16_t* vtb = (bf16_t*)(ws + off); off += (size_t)BH_ * SEQ_ * HD_ * 2;  // 8 MB
  bf16_t* atb = (bf16_t*)(ws + off); off += (size_t)M_TOT * EMB_ * 2;      // 8 MB

  cvt_f32_bf16<<<(M_TOT * EMB_) / 256, 256, 0, stream>>>(x, xb, M_TOT * EMB_);
  cvt_f32_bf16<<<(3 * EMB_ * EMB_) / 256, 256, 0, stream>>>(wqkv, wqb, 3 * EMB_ * EMB_);
  cvt_f32_bf16<<<(EMB_ * EMB_) / 256, 256, 0, stream>>>(wout, wob, EMB_ * EMB_);

  // QKV: M=4096, N=3072, K=1024
  gemm_bf16_128x128<1, 3 * EMB_, EMB_>
      <<<dim3(3 * EMB_ / 128, M_TOT / 128), 256, 0, stream>>>(
          xb, wqb, bqkv, nullptr, qb, kb, vtb);

  // 32 q-groups x 32 heads; block = 4 waves on one head
  flash_attn<<<dim3(SEQ_ / 64, BH_), 128, 0, stream>>>(qb, kb, vtb, atb);

  // Out projection: M=4096, N=1024, K=1024 -> fp32 d_out
  gemm_bf16_128x128<0, EMB_, EMB_>
      <<<dim3(EMB_ / 128, M_TOT / 128), 256, 0, stream>>>(
          atb, wob, bout, out, nullptr, nullptr, nullptr);
}
